// CausalSelfAttention_29746943492154
// MI455X (gfx1250) — compile-verified, matrix operand order checked by reading, not run on hardware
//
#include <hip/hip_runtime.h>
#include <hip/hip_bf16.h>

typedef __attribute__((ext_vector_type(16))) _Float16 v16h;
typedef __attribute__((ext_vector_type(8)))  float    v8f;

#define B_  2
#define T_  2048
#define C_  1024
#define H_  16
#define D_  64
#define C3_ 3072
#define BT_ (B_ * T_)

static __device__ __forceinline__ v8f wmma16(v16h a, v16h b, v8f c) {
    // D = A(16x32 f16) * B(32x16 f16) + C(16x16 f32)
    return __builtin_amdgcn_wmma_f32_16x16x32_f16(false, a, false, b, (short)0, c, false, false);
}

static __device__ __forceinline__ float rowmax16(float v) {
    v = fmaxf(v, __shfl_xor(v, 1, 32));
    v = fmaxf(v, __shfl_xor(v, 2, 32));
    v = fmaxf(v, __shfl_xor(v, 4, 32));
    v = fmaxf(v, __shfl_xor(v, 8, 32));
    return v;
}
static __device__ __forceinline__ float rowsum16(float v) {
    v += __shfl_xor(v, 1, 32);
    v += __shfl_xor(v, 2, 32);
    v += __shfl_xor(v, 4, 32);
    v += __shfl_xor(v, 8, 32);
    return v;
}

// ---------------- f32 -> f16 cast ----------------
__global__ void cast_f16_kernel(const float* __restrict__ in, _Float16* __restrict__ out, int n) {
    int i = blockIdx.x * blockDim.x + threadIdx.x;
    if (i < n) out[i] = (_Float16)in[i];
}

// in: [K rows x N cols] f32 row-major;  out: [N x K] f16 (i.e. transposed)
__global__ void cast_f16_T_kernel(const float* __restrict__ in, _Float16* __restrict__ out, int K, int N) {
    int i = blockIdx.x * blockDim.x + threadIdx.x;
    if (i < K * N) {
        int n = i / K;
        int k = i - n * K;
        out[i] = (_Float16)in[k * N + n];
    }
}

// ---------------- QKV GEMM: qkv = x @ w_attn + b_attn, scatter to q/k/vT ----------------
// grid (C3/128, BT/64), block 128 (4 waves). Wave -> 16 rows x 128 cols tile.
__global__ __launch_bounds__(128) void qkv_gemm_kernel(
    const _Float16* __restrict__ X,      // [BT, C] f16
    const _Float16* __restrict__ Wt,     // [3C, C] f16 (w_attn transposed)
    const float*    __restrict__ bias,   // [3C]
    _Float16* __restrict__ Q,            // [B,H,T,D]
    _Float16* __restrict__ Kq,           // [B,H,T,D]
    _Float16* __restrict__ Vt)           // [B,H,D,T]
{
    const int lane  = threadIdx.x & 31;
    const int wave  = threadIdx.x >> 5;
    const int mbase = blockIdx.y * 64 + wave * 16;
    const int nbase = blockIdx.x * 128;
    const int nloc  = lane & 15;
    const int g16   = (lane >> 4) * 16;   // free K-slot offset (same for A and B)

    const _Float16* arow = X + (mbase + nloc) * C_ + g16;
    const _Float16* brow[8];
#pragma unroll
    for (int j = 0; j < 8; ++j)
        brow[j] = Wt + (nbase + j * 16 + nloc) * C_ + g16;

    v8f acc[8] = {};
    for (int kk = 0; kk < C_; kk += 32) {
        // preload all operands for this K step...
        v16h a = *(const v16h*)(arow + kk);
        v16h breg[8];
#pragma unroll
        for (int j = 0; j < 8; ++j)
            breg[j] = *(const v16h*)(brow[j] + kk);
        __builtin_prefetch(arow + kk + 128, 0, 3);   // near-scope global_prefetch_b8
        // ...then keep them live: loads may not sink below, WMMAs may not hoist above
        __builtin_amdgcn_sched_barrier(0);
#pragma unroll
        for (int j = 0; j < 8; ++j)
            acc[j] = wmma16(a, breg[j], acc[j]);
    }

    const int rowoff = (lane >> 4) * 8;
#pragma unroll
    for (int j = 0; j < 8; ++j) {
        const int col = nbase + j * 16 + nloc;   // 0..3071
        const float bv = bias[col];
        const int seg = col >> 10;               // 0=q 1=k 2=v
        const int c   = col & (C_ - 1);
        const int h   = c >> 6;
        const int d   = c & 63;
#pragma unroll
        for (int r = 0; r < 8; ++r) {
            const int trow = mbase + r + rowoff; // 0..BT-1
            const int b = trow >> 11;            // /T
            const int t = trow & (T_ - 1);
            const _Float16 hv = (_Float16)(acc[j][r] + bv);
            if (seg == 0)      Q [(((b * H_) + h) * T_ + t) * D_ + d] = hv;
            else if (seg == 1) Kq[(((b * H_) + h) * T_ + t) * D_ + d] = hv;
            else               Vt[(((b * H_) + h) * D_ + d) * T_ + t] = hv;
        }
    }
}

// ---------------- Flash attention: per wave, one 16-row Q tile ----------------
// grid (T/16/4, B*H), block 128 (4 waves).
__global__ __launch_bounds__(128) void attn_kernel(
    const _Float16* __restrict__ Q,   // [B,H,T,D]
    const _Float16* __restrict__ Kq,  // [B,H,T,D]
    const _Float16* __restrict__ Vt,  // [B,H,D,T]
    _Float16* __restrict__ Y)         // [BT, C] f16
{
    __shared__ __align__(32) _Float16 plds[4][16 * 32];

    const int lane  = threadIdx.x & 31;
    const int wave  = threadIdx.x >> 5;
    const int qtile = blockIdx.x * 4 + wave;   // 0..127
    const int bh    = blockIdx.y;              // 0..31
    const int b     = bh >> 4;
    const int h     = bh & 15;
    const int tbase = qtile * 16;
    const int nloc  = lane & 15;
    const int g16   = (lane >> 4) * 16;
    const int rowoff = (lane >> 4) * 8;

    // A operand for Q·Kᵀ: lane row = tbase + nloc, D split in two K=32 halves
    const _Float16* qrow = Q + ((size_t)bh * T_ + tbase + nloc) * D_;
    const v16h a0 = *(const v16h*)(qrow + g16);
    const v16h a1 = *(const v16h*)(qrow + 32 + g16);

    v8f o0 = {}, o1 = {}, o2 = {}, o3 = {};
    float m_run[8], l_run[8];
#pragma unroll
    for (int r = 0; r < 8; ++r) { m_run[r] = -1e30f; l_run[r] = 0.0f; }

    _Float16* pl = &plds[wave][0];
    const int nchunks = (qtile >> 1) + 1;      // ceil((qtile+1)/2) 32-wide chunks

    for (int ch = 0; ch < nchunks; ++ch) {
        const int s0 = ch * 32;
        // speculative near-scope prefetch of next chunk's K and V lines
        __builtin_prefetch(Kq + ((size_t)bh * T_ + s0 + 32 + nloc) * D_ + g16, 0, 3);
        __builtin_prefetch(Vt + ((size_t)bh * D_ + nloc) * T_ + s0 + 32 + g16, 0, 3);

        // preload both K tiles AND the V tiles for this chunk, then issue WMMAs grouped
        const _Float16* krow0 = Kq + ((size_t)bh * T_ + s0 + nloc) * D_;
        const _Float16* krow1 = Kq + ((size_t)bh * T_ + s0 + 16 + nloc) * D_;
        v16h kb00 = *(const v16h*)(krow0 + g16);
        v16h kb01 = *(const v16h*)(krow0 + 32 + g16);
        v16h kb10 = *(const v16h*)(krow1 + g16);
        v16h kb11 = *(const v16h*)(krow1 + 32 + g16);
        const _Float16* vbase = Vt + (size_t)bh * D_ * T_ + s0 + g16;
        v16h vb0 = *(const v16h*)(vbase + (size_t)(0  + nloc) * T_);
        v16h vb1 = *(const v16h*)(vbase + (size_t)(16 + nloc) * T_);
        v16h vb2 = *(const v16h*)(vbase + (size_t)(32 + nloc) * T_);
        v16h vb3 = *(const v16h*)(vbase + (size_t)(48 + nloc) * T_);
        __builtin_amdgcn_sched_barrier(0);

        v8f st[2];
        st[0] = wmma16(a1, kb01, wmma16(a0, kb00, (v8f){}));
        st[1] = wmma16(a1, kb11, wmma16(a0, kb10, (v8f){}));

#pragma unroll
        for (int sti = 0; sti < 2; ++sti) {
            const int scol = s0 + sti * 16 + nloc;
#pragma unroll
            for (int r = 0; r < 8; ++r) {
                const int trow = tbase + r + rowoff;
                float v = st[sti][r] * 0.125f;            // 1/sqrt(64)
                st[sti][r] = (scol <= trow) ? v : -1e30f; // causal mask
            }
        }

        // online softmax update (row = vgpr index, stats are elementwise)
#pragma unroll
        for (int r = 0; r < 8; ++r) {
            float mx = rowmax16(fmaxf(st[0][r], st[1][r]));
            float nm = fmaxf(m_run[r], mx);
            float alpha = __expf(m_run[r] - nm);
            m_run[r] = nm;
            float p0 = __expf(st[0][r] - nm);
            float p1 = __expf(st[1][r] - nm);
            l_run[r] = l_run[r] * alpha + rowsum16(p0 + p1);
            o0[r] *= alpha; o1[r] *= alpha; o2[r] *= alpha; o3[r] *= alpha;
            // stage P (f16) into LDS in row-major [16 x 32]
            const int m = r + rowoff;
            pl[m * 32 + nloc]      = (_Float16)p0;
            pl[m * 32 + 16 + nloc] = (_Float16)p1;
        }

        // reload P as A operand (same free K-slot mapping as V's B operand)
        v16h pa = *(const v16h*)(pl + nloc * 32 + g16);
        o0 = wmma16(pa, vb0, o0);
        o1 = wmma16(pa, vb1, o1);
        o2 = wmma16(pa, vb2, o2);
        o3 = wmma16(pa, vb3, o3);
    }

    // finalize: divide by l, write y[b, t, h*64 + d] as f16
    v8f oo[4] = {o0, o1, o2, o3};
#pragma unroll
    for (int r = 0; r < 8; ++r) {
        const float inv = 1.0f / l_run[r];
        const int t = tbase + r + rowoff;
        _Float16* yrow = Y + ((size_t)b * T_ + t) * C_ + h * D_;
#pragma unroll
        for (int j = 0; j < 4; ++j)
            yrow[j * 16 + nloc] = (_Float16)(oo[j][r] * inv);
    }
}

// ---------------- Output projection: out = y @ w_proj + b_proj ----------------
// grid (C/128, BT/64), block 128. Wave -> 16 rows x 128 cols tile.
__global__ __launch_bounds__(128) void proj_gemm_kernel(
    const _Float16* __restrict__ Yv,    // [BT, C] f16
    const _Float16* __restrict__ Wt,    // [C, C] f16 (w_proj transposed)
    const float*    __restrict__ bias,  // [C]
    float* __restrict__ Out)            // [BT, C] f32
{
    const int lane  = threadIdx.x & 31;
    const int wave  = threadIdx.x >> 5;
    const int mbase = blockIdx.y * 64 + wave * 16;
    const int nbase = blockIdx.x * 128;
    const int nloc  = lane & 15;
    const int g16   = (lane >> 4) * 16;

    const _Float16* arow = Yv + (mbase + nloc) * C_ + g16;
    const _Float16* brow[8];
#pragma unroll
    for (int j = 0; j < 8; ++j)
        brow[j] = Wt + (nbase + j * 16 + nloc) * C_ + g16;

    v8f acc[8] = {};
    for (int kk = 0; kk < C_; kk += 32) {
        v16h a = *(const v16h*)(arow + kk);
        v16h breg[8];
#pragma unroll
        for (int j = 0; j < 8; ++j)
            breg[j] = *(const v16h*)(brow[j] + kk);
        __builtin_prefetch(arow + kk + 128, 0, 3);
        __builtin_amdgcn_sched_barrier(0);
#pragma unroll
        for (int j = 0; j < 8; ++j)
            acc[j] = wmma16(a, breg[j], acc[j]);
    }

    const int rowoff = (lane >> 4) * 8;
#pragma unroll
    for (int j = 0; j < 8; ++j) {
        const int col = nbase + j * 16 + nloc;
        const float bv = bias[col];
#pragma unroll
        for (int r = 0; r < 8; ++r) {
            const int trow = mbase + r + rowoff;
            Out[(size_t)trow * C_ + col] = acc[j][r] + bv;
        }
    }
}

extern "C" void kernel_launch(void* const* d_in, const int* in_sizes, int n_in,
                              void* d_out, int out_size, void* d_ws, size_t ws_size,
                              hipStream_t stream) {
    const float* x      = (const float*)d_in[0];
    const float* w_attn = (const float*)d_in[1];
    const float* b_attn = (const float*)d_in[2];
    const float* w_proj = (const float*)d_in[3];
    const float* b_proj = (const float*)d_in[4];
    float* out = (float*)d_out;

    // workspace carve-up (f16 elements); total ~48 MB
    _Float16* ws = (_Float16*)d_ws;
    _Float16* x16  = ws;                               // BT*C
    _Float16* waT  = x16  + (size_t)BT_ * C_;          // 3C*C
    _Float16* wpT  = waT  + (size_t)C3_ * C_;          // C*C
    _Float16* q16  = wpT  + (size_t)C_  * C_;          // B*H*T*D
    _Float16* k16  = q16  + (size_t)B_ * H_ * T_ * D_;
    _Float16* vT16 = k16  + (size_t)B_ * H_ * T_ * D_;
    _Float16* y16  = vT16 + (size_t)B_ * H_ * T_ * D_;

    const int nx = BT_ * C_;
    cast_f16_kernel<<<(nx + 255) / 256, 256, 0, stream>>>(x, x16, nx);

    const int nwa = C_ * C3_;
    cast_f16_T_kernel<<<(nwa + 255) / 256, 256, 0, stream>>>(w_attn, waT, C_, C3_);
    const int nwp = C_ * C_;
    cast_f16_T_kernel<<<(nwp + 255) / 256, 256, 0, stream>>>(w_proj, wpT, C_, C_);

    qkv_gemm_kernel<<<dim3(C3_ / 128, BT_ / 64), 128, 0, stream>>>(x16, waT, b_attn, q16, k16, vT16);

    attn_kernel<<<dim3(T_ / 16 / 4, B_ * H_), 128, 0, stream>>>(q16, k16, vT16, y16);

    proj_gemm_kernel<<<dim3(C_ / 128, BT_ / 64), 128, 0, stream>>>(y16, wpT, b_proj, out);
}
